// Autoencoder_17566416241003
// MI455X (gfx1250) — matrix operand order repaired and checked by reading, not compiled, not run on hardware
//
#include <hip/hip_runtime.h>

typedef __bf16 bf16_t;
typedef __attribute__((ext_vector_type(16))) __bf16 v16bf;
typedef __attribute__((ext_vector_type(8)))  __bf16 v8bf;
typedef __attribute__((ext_vector_type(8)))  float  v8f;

#define F 256
#define NEG_INF (-3.402823466e38f)

// Load a 16-element bf16 A/B fragment from two contiguous 16-byte chunks.
__device__ __forceinline__ v16bf load_frag16(const bf16_t* p0, const bf16_t* p1) {
    union { v16bf v; v8bf h[2]; } u;
    u.h[0] = *(const v8bf*)p0;
    u.h[1] = *(const v8bf*)p1;
    return u.v;
}

// ---------------- elementwise helpers ----------------

__global__ void k_f32_to_bf16(const float* __restrict__ in, bf16_t* __restrict__ out, int n) {
    int i = blockIdx.x * blockDim.x + threadIdx.x;
    int stride = gridDim.x * blockDim.x;
    for (; i < n; i += stride) out[i] = (bf16_t)in[i];
}

// dst[n*256+k] = bf16(src[k*256+n])   (256x256 transpose+convert)
__global__ void k_transpose_bf16(const float* __restrict__ src, bf16_t* __restrict__ dst) {
    int i = blockIdx.x * blockDim.x + threadIdx.x;  // 65536 threads total
    int n = i >> 8, k = i & 255;
    dst[(size_t)n * F + k] = (bf16_t)src[(size_t)k * F + n];
}

__global__ void k_fill(float* __restrict__ p, float v, int n) {
    int i = blockIdx.x * blockDim.x + threadIdx.x;
    int stride = gridDim.x * blockDim.x;
    for (; i < n; i += stride) p[i] = v;
}

// h_bf16 = bf16(relu(agg + bias[n]))  ; -FLT_MAX sentinel (no in-edges) -> 0
__global__ void k_finalize(const float* __restrict__ agg, const float* __restrict__ bias,
                           bf16_t* __restrict__ out, int total) {
    int i = blockIdx.x * blockDim.x + threadIdx.x;
    int stride = gridDim.x * blockDim.x;
    for (; i < total; i += stride) {
        int n = i & (F - 1);
        out[i] = (bf16_t)fmaxf(agg[i] + bias[n], 0.0f);
    }
}

// ---------------- node-level GEMM: C[M,256] = A[M,256] @ Bt^T + bias ----------------
// A: bf16 row-major [M,256]; Bt: bf16 [N=256 rows][K=256] (i.e. W transposed)
// MODE 0: fp32 out, no relu.  MODE 1: bf16 out with relu.
template <int MODE>
__global__ void __launch_bounds__(512)
k_gemm256(const bf16_t* __restrict__ A, const bf16_t* __restrict__ Bt,
          const float* __restrict__ bias, float* __restrict__ outF,
          bf16_t* __restrict__ outB) {
    const int wave = threadIdx.x >> 5;        // 16 waves: one 16-col tile each
    const int lane = threadIdx.x & 31;
    const int i0   = blockIdx.x << 4;         // 16 rows per block
    const int n0   = wave << 4;
    const int hi   = lane >> 4;

    const bf16_t* arow = A  + (size_t)(i0 + (lane & 15)) * F + hi * 8;
    const bf16_t* brow = Bt + (size_t)(n0 + (lane & 15)) * F + hi * 8;

    v8f acc = {};
#pragma unroll
    for (int kk = 0; kk < 8; ++kk) {
        v16bf a = load_frag16(arow + kk * 32, arow + kk * 32 + 16);
        v16bf b = load_frag16(brow + kk * 32, brow + kk * 32 + 16);
        acc = __builtin_amdgcn_wmma_f32_16x16x32_bf16(false, a, false, b, (short)0, acc,
                                                      false, false);
    }

    const int   n  = n0 + (lane & 15);
    const float bn = bias[n];
#pragma unroll
    for (int r = 0; r < 8; ++r) {
        int    m   = r + (hi << 3);
        float  v   = acc[r] + bn;
        size_t idx = (size_t)(i0 + m) * F + n;
        if (MODE == 0) outF[idx] = v;
        else           outB[idx] = (bf16_t)fmaxf(v, 0.0f);
    }
}

// ---------------- per-edge message GEMM + max-scatter ----------------
// For each group of 16 edges:
//   T[e][:] = relu(pre[src_e] + (pos[src_e]-pos[dst_e]) @ wpos)   (built in LDS, bf16)
//   M = T @ W_b   (16x256 = 8 WMMA per wave-tile)
//   agg[dst_e][:] = atomic float max
__global__ void __launch_bounds__(512)
k_edge(const float* __restrict__ pre, const float* __restrict__ pos,
       const long long* __restrict__ srcA, const long long* __restrict__ dstA,
       const float* __restrict__ wpos,      // 3 x 256 fp32 (rows 256..258 of wA)
       const bf16_t* __restrict__ Bt,       // W_b transposed, bf16 [256][256]
       float* __restrict__ agg, int nGroups) {
    __shared__ bf16_t T[16][264];           // 264 = 256 + 8 pad -> conflict-free ds_load_b128
    __shared__ int    sD[16];

    const int tid  = threadIdx.x;
    const int wave = tid >> 5;
    const int lane = tid & 31;
    const int n0   = wave << 4;
    const int hi   = lane >> 4;

    // B fragments resident in registers across the whole grid-stride loop
    const bf16_t* brow = Bt + (size_t)(n0 + (lane & 15)) * F + hi * 8;
    v16bf bfrag[8];
#pragma unroll
    for (int kk = 0; kk < 8; ++kk)
        bfrag[kk] = load_frag16(brow + kk * 32, brow + kk * 32 + 16);

    const bf16_t* abase = &T[lane & 15][0] + hi * 8;

    for (int g = blockIdx.x; g < nGroups; g += gridDim.x) {
        {   // build message tile: thread -> (edge e, 8 features at column c)
            const int e = tid >> 5;
            const int c = (tid & 31) << 3;
            const long long s = srcA[(size_t)g * 16 + e];
            const long long d = dstA[(size_t)g * 16 + e];
            if ((tid & 31) == 0) sD[e] = (int)d;
            const float dpx = pos[s * 3 + 0] - pos[d * 3 + 0];
            const float dpy = pos[s * 3 + 1] - pos[d * 3 + 1];
            const float dpz = pos[s * 3 + 2] - pos[d * 3 + 2];
            const float* pr = pre + (size_t)s * F + c;
#pragma unroll
            for (int j = 0; j < 8; ++j) {
                int   n = c + j;
                float v = pr[j] + dpx * wpos[n] + dpy * wpos[F + n] + dpz * wpos[2 * F + n];
                T[e][n] = (bf16_t)fmaxf(v, 0.0f);
            }
        }
        __syncthreads();

        v8f acc = {};
#pragma unroll
        for (int kk = 0; kk < 8; ++kk) {
            v16bf a = load_frag16(abase + kk * 32, abase + kk * 32 + 16);
            acc = __builtin_amdgcn_wmma_f32_16x16x32_bf16(false, a, false, bfrag[kk],
                                                          (short)0, acc, false, false);
        }

        const int n = n0 + (lane & 15);
#pragma unroll
        for (int r = 0; r < 8; ++r) {
            int    m    = r + (hi << 3);             // edge index within group
            float* addr = agg + (size_t)sD[m] * F + n;
            __hip_atomic_fetch_max(addr, acc[r], __ATOMIC_RELAXED, __HIP_MEMORY_SCOPE_AGENT);
        }
        __syncthreads();
    }
}

// ---------------- launch ----------------

extern "C" void kernel_launch(void* const* d_in, const int* in_sizes, int n_in,
                              void* d_out, int out_size, void* d_ws, size_t ws_size,
                              hipStream_t stream) {
    const float*     x    = (const float*)d_in[0];
    const float*     pos  = (const float*)d_in[1];
    const long long* eidx = (const long long*)d_in[2];
    const float *w1a = (const float*)d_in[3],  *b1a = (const float*)d_in[4];
    const float *w1b = (const float*)d_in[5],  *b1b = (const float*)d_in[6];
    const float *w2a = (const float*)d_in[7],  *b2a = (const float*)d_in[8];
    const float *w2b = (const float*)d_in[9],  *b2b = (const float*)d_in[10];
    const float *wd1 = (const float*)d_in[11], *bd1 = (const float*)d_in[12];
    const float *wd2 = (const float*)d_in[13], *bd2 = (const float*)d_in[14];

    const int       N       = in_sizes[0] / F;           // 50000
    const long long E       = in_sizes[2] / 2;           // 800000
    const int       nGroups = (int)(E / 16);             // 50000
    const long long* srcA = eidx;
    const long long* dstA = eidx + E;

    // workspace carve-out
    char*  ws  = (char*)d_ws;
    size_t off = 0;
    auto carve = [&](size_t bytes) -> void* {
        void* p = ws + off;
        off = (off + bytes + 255) & ~(size_t)255;
        return p;
    };
    const size_t NF = (size_t)N * F;
    bf16_t* xb   = (bf16_t*)carve(NF * 2);   // x bf16; reused as decoder hidden z
    bf16_t* hb   = (bf16_t*)carve(NF * 2);   // h1 bf16, then h2 bf16
    float*  pre  = (float*) carve(NF * 4);   // per-node precomputed first-GEMM term
    float*  agg  = (float*) carve(NF * 4);   // segment-max accumulator
    bf16_t* w1aT = (bf16_t*)carve((size_t)F * F * 2);
    bf16_t* w1bT = (bf16_t*)carve((size_t)F * F * 2);
    bf16_t* w2aT = (bf16_t*)carve((size_t)F * F * 2);
    bf16_t* w2bT = (bf16_t*)carve((size_t)F * F * 2);
    bf16_t* wd1T = (bf16_t*)carve((size_t)F * F * 2);
    bf16_t* wd2T = (bf16_t*)carve((size_t)F * F * 2);

    const int EW = 256;                   // elementwise block
    const int GB = N / 16;                // gemm blocks (N divisible by 16)

    // prep: convert / transpose
    k_f32_to_bf16<<<2048, EW, 0, stream>>>(x, xb, (int)NF);
    k_transpose_bf16<<<256, EW, 0, stream>>>(w1a, w1aT);  // feature rows 0..255 only
    k_transpose_bf16<<<256, EW, 0, stream>>>(w1b, w1bT);
    k_transpose_bf16<<<256, EW, 0, stream>>>(w2a, w2aT);
    k_transpose_bf16<<<256, EW, 0, stream>>>(w2b, w2bT);
    k_transpose_bf16<<<256, EW, 0, stream>>>(wd1, wd1T);
    k_transpose_bf16<<<256, EW, 0, stream>>>(wd2, wd2T);

    // ---- layer 1 ----
    k_fill<<<2048, EW, 0, stream>>>(agg, NEG_INF, (int)NF);
    k_gemm256<0><<<GB, 512, 0, stream>>>(xb, w1aT, b1a, pre, nullptr);
    k_edge<<<2048, 512, 0, stream>>>(pre, pos, srcA, dstA, w1a + (size_t)256 * F, w1bT,
                                     agg, nGroups);
    k_finalize<<<2048, EW, 0, stream>>>(agg, b1b, hb, (int)NF);

    // ---- layer 2 ----
    k_fill<<<2048, EW, 0, stream>>>(agg, NEG_INF, (int)NF);
    k_gemm256<0><<<GB, 512, 0, stream>>>(hb, w2aT, b2a, pre, nullptr);
    k_edge<<<2048, 512, 0, stream>>>(pre, pos, srcA, dstA, w2a + (size_t)256 * F, w2bT,
                                     agg, nGroups);
    k_finalize<<<2048, EW, 0, stream>>>(agg, b2b, hb, (int)NF);

    // ---- decoder ----
    k_gemm256<1><<<GB, 512, 0, stream>>>(hb, wd1T, bd1, nullptr, xb);       // z = relu(h@wd1+bd1)
    k_gemm256<0><<<GB, 512, 0, stream>>>(xb, wd2T, bd2, (float*)d_out, nullptr);
}
